// SelfAttention_8452495638749
// MI455X (gfx1250) — matrix operand order repaired and checked by reading, not compiled, not run on hardware
//
#include <hip/hip_runtime.h>
#include <hip/hip_bf16.h>

// ---------------------------------------------------------------------------
// Self-attention, MI455X (gfx1250), wave32 + v_wmma_f32_16x16x32_bf16.
// B=8, S=2048, D=1024.  fp32 inputs cast once to bf16; all five GEMMs
// (Q,K,V projections, QK^T, PV) run on the WMMA pipe with f32 accumulators.
// WMMA loops are software-pipelined with *2x-unrolled ping-pong fragment
// buffers*: no loop-carried register rotation -> no v_mov copies and no
// WMMA WAR-hazard v_nops; loads for kc+2/kc+3 overlap the WMMAs of kc/kc+1.
// Softmax is fused: each workgroup holds 16 full score rows (16x2048 f32 =
// 128KB) + bf16 probs (64KB) in CDNA5's 320KB LDS, avoiding a 128MB HBM
// round-trip for the score matrix.
// Workspace layout (needs 134MB):
//   [  0MB) Xbf   16384x1024 bf16
//   [ 32MB) WtQ   1024x1024 bf16  (W transposed -> [out][in])
//   [ 34MB) WtK
//   [ 36MB) WtV
//   [ 38MB) Qbf   [8][2048][1024] bf16 row-major
//   [ 70MB) Kbf   [8][2048][1024] bf16 row-major  (acts as Bt for Q@K^T)
//   [102MB) Vt    [8][1024][2048] bf16            (acts as Bt for P@V)
// ---------------------------------------------------------------------------

typedef __bf16 bf16;
typedef __attribute__((ext_vector_type(4)))  __bf16 bf16x4;
typedef __attribute__((ext_vector_type(8)))  __bf16 bf16x8;
typedef __attribute__((ext_vector_type(16))) __bf16 bf16x16;
typedef __attribute__((ext_vector_type(8)))  float  f32x8;

#define BATCH  8
#define SEQ    2048
#define DMODEL 1024
#define ROWS   (BATCH * SEQ)
#define MQ     16            // query rows per attention workgroup

// ---- WMMA fragment loaders (contiguous k order k = 16*(lane/16) + s; A and B agree) ----------
// A (16x32 bf16, row-major source): lane m in 0..15 -> row m, K 0..15;
// lanes 16..31 -> same rows, K 16..31.
__device__ __forceinline__ bf16x16 load_frag_A(const bf16* base, int stride, int lane) {
  int m  = lane & 15;
  int a0 = (lane < 16) ? 0 : 16;
  const bf16* p = base + (size_t)m * stride + a0;
  union { bf16x16 v; bf16x8 h[2]; } u;
  u.h[0] = *(const bf16x8*)(p);
  u.h[1] = *(const bf16x8*)(p + 8);
  return u.v;
}

// B (32x16 bf16) loaded from *transposed* storage Bt[n][k]: lane n in 0..15 ->
// col n, K rows 0..15; lanes 16..31 -> K rows 16..31.  32B contiguous / lane.
__device__ __forceinline__ bf16x16 load_frag_B(const bf16* baseT, int stride, int lane) {
  int n  = lane & 15;
  int kb = (lane < 16) ? 0 : 16;
  return *(const bf16x16*)(baseT + (size_t)n * stride + kb);
}

__device__ __forceinline__ f32x8 wmma_bf16(bf16x16 a, bf16x16 b, f32x8 c) {
  return __builtin_amdgcn_wmma_f32_16x16x32_bf16(false, a, false, b, (short)0, c,
                                                 false, false);
}

// ---- fp32 -> bf16 casts ----------------------------------------------------
__global__ __launch_bounds__(256) void cast_x_kernel(const float* __restrict__ x,
                                                     bf16* __restrict__ xb) {
  int i = (blockIdx.x * 256 + threadIdx.x) * 4;
  float4 v = *(const float4*)(x + i);
  bf16x4 o;
  o.x = (bf16)v.x; o.y = (bf16)v.y; o.z = (bf16)v.z; o.w = (bf16)v.w;
  *(bf16x4*)(xb + i) = o;
}

// W stored [in][out]; write Wt[out][in] bf16.
__global__ __launch_bounds__(256) void cast_w_kernel(const float* __restrict__ W,
                                                     bf16* __restrict__ Wt) {
  int idx = blockIdx.x * 256 + threadIdx.x;   // 0 .. 1M-1
  int i = idx >> 10;                          // in
  int o = idx & 1023;                         // out
  Wt[(size_t)o * DMODEL + i] = (bf16)W[idx];
}

// ===========================================================================
// QKV projection: Y = Xbf @ Wt^T + bias, output bf16.
// Wave-tile: M=32, N=64 (2x4 accumulators).  8 waves / block.
// ===========================================================================
struct GemmFrags { bf16x16 a[2]; bf16x16 b[4]; };

__device__ __forceinline__ void gemm_load(GemmFrags& f, const bf16* Ab,
                                          const bf16* Bb, int kc, int lane) {
#pragma unroll
  for (int mi = 0; mi < 2; ++mi)
    f.a[mi] = load_frag_A(Ab + (size_t)mi * 16 * DMODEL + kc * 32, DMODEL, lane);
#pragma unroll
  for (int t = 0; t < 4; ++t)
    f.b[t] = load_frag_B(Bb + (size_t)t * 16 * DMODEL + kc * 32, DMODEL, lane);
}

__device__ __forceinline__ void gemm_mma(f32x8 acc[2][4], const GemmFrags& f) {
#pragma unroll
  for (int mi = 0; mi < 2; ++mi)
#pragma unroll
    for (int t = 0; t < 4; ++t)
      acc[mi][t] = wmma_bf16(f.a[mi], f.b[t], acc[mi][t]);
}

__global__ __launch_bounds__(256)
void qkv_gemm_kernel(const bf16* __restrict__ X,    // [ROWS][DMODEL]
                     const bf16* __restrict__ Wt,   // [DMODEL(out)][DMODEL(in)]
                     const float* __restrict__ bias,
                     bf16* __restrict__ out,
                     int transpose_out) {            // 0: [ROWS][D]; 1: [B][D][S]
  int lane = threadIdx.x & 31;
  int waveGlobal = blockIdx.x * 8 + (threadIdx.x >> 5);
  const int tilesN = DMODEL / 64;                   // 16
  int m0 = (waveGlobal / tilesN) * 32;
  int n0 = (waveGlobal % tilesN) * 64;

  const bf16* Ab = X  + (size_t)m0 * DMODEL;
  const bf16* Bb = Wt + (size_t)n0 * DMODEL;

  f32x8 acc[2][4] = {};
  GemmFrags f0, f1;                 // ping-pong buffers (no rotation copies)
  gemm_load(f0, Ab, Bb, 0, lane);
  gemm_load(f1, Ab, Bb, 1, lane);

  for (int kc = 0; kc < DMODEL / 32 - 2; kc += 2) {
    gemm_mma(acc, f0);              // consume kc
    gemm_load(f0, Ab, Bb, kc + 2, lane);
    gemm_mma(acc, f1);              // consume kc+1 (overlaps f0 loads)
    gemm_load(f1, Ab, Bb, kc + 3, lane);
  }
  gemm_mma(acc, f0);                // kc = K-2
  gemm_mma(acc, f1);                // kc = K-1

  int nLane = lane & 15;
  int mBase = (lane < 16) ? 0 : 8;
#pragma unroll
  for (int mi = 0; mi < 2; ++mi) {
#pragma unroll
    for (int t = 0; t < 4; ++t) {
      int n = n0 + t * 16 + nLane;
      float bv = bias[n];
#pragma unroll
      for (int r = 0; r < 8; ++r) {
        int m = m0 + mi * 16 + mBase + r;
        float v = acc[mi][t][r] + bv;
        if (!transpose_out) {
          out[(size_t)m * DMODEL + n] = (bf16)v;
        } else {
          int bb = m / SEQ, s = m % SEQ;
          out[(size_t)bb * DMODEL * SEQ + (size_t)n * SEQ + s] = (bf16)v;
        }
      }
    }
  }
}

// ===========================================================================
// Fused attention: per workgroup = 16 query rows, full-row softmax in LDS.
// ===========================================================================
struct SFrags { bf16x16 a; bf16x16 b[4]; };

__device__ __forceinline__ void s_load(SFrags& f, const bf16* Qb, const bf16* Kb,
                                       int n0, int kc, int lane) {
  f.a = load_frag_A(Qb + kc * 32, DMODEL, lane);
#pragma unroll
  for (int t = 0; t < 4; ++t)
    f.b[t] = load_frag_B(Kb + (size_t)(n0 + t * 16) * DMODEL + kc * 32, DMODEL, lane);
}

__device__ __forceinline__ void s_mma(f32x8 acc[4], const SFrags& f) {
#pragma unroll
  for (int t = 0; t < 4; ++t) acc[t] = wmma_bf16(f.a, f.b[t], acc[t]);
}

// One k-chunk of O = P@V for 16 d-tiles; the b-fragment chain over t is
// fully unrolled SSA (no carried copies).
__device__ __forceinline__ void pv_step(f32x8* acc, bf16x16 a, bf16x16 b0,
                                        const bf16* Vb, int wave, int kc, int lane) {
  bf16x16 bc = b0;
#pragma unroll
  for (int t = 0; t < 16; ++t) {
    bf16x16 bn = bc;
    if (t < 15)
      bn = load_frag_B(Vb + (size_t)((wave * 16 + t + 1) * 16) * SEQ + kc * 32,
                       SEQ, lane);
    acc[t] = wmma_bf16(a, bc, acc[t]);
    bc = bn;
  }
}

__global__ __launch_bounds__(128)
void attn_kernel(const bf16* __restrict__ Q,   // [B][S][D]
                 const bf16* __restrict__ K,   // [B][S][D]
                 const bf16* __restrict__ Vt,  // [B][D][S]
                 float* __restrict__ O) {      // [B][S][D]
  extern __shared__ char smem[];
  float* Ssc = (float*)smem;                       // [MQ][SEQ]  128KB
  bf16*  P   = (bf16*)(smem + MQ * SEQ * 4);       // [MQ][SEQ]   64KB

  int lane = threadIdx.x & 31;
  int wave = threadIdx.x >> 5;                     // 0..3
  int b    = blockIdx.x / (SEQ / MQ);
  int m0   = (blockIdx.x % (SEQ / MQ)) * MQ;

  const bf16* Qb = Q  + (size_t)b * SEQ * DMODEL + (size_t)m0 * DMODEL;
  const bf16* Kb = K  + (size_t)b * SEQ * DMODEL;
  const bf16* Vb = Vt + (size_t)b * DMODEL * SEQ;

  // ---- S = scale * Q @ K^T  (K row-major == Bt layout) ----
  // Wave owns 32 key tiles, processed in 8 groups of 4 sharing the Q fragment.
  for (int g = 0; g < 8; ++g) {
    int n0 = (wave * 32 + g * 4) * 16;
    f32x8 acc[4] = {};
    SFrags f0, f1;
    s_load(f0, Qb, Kb, n0, 0, lane);
    s_load(f1, Qb, Kb, n0, 1, lane);
    for (int kc = 0; kc < DMODEL / 32 - 2; kc += 2) {
      s_mma(acc, f0);
      s_load(f0, Qb, Kb, n0, kc + 2, lane);
      s_mma(acc, f1);
      s_load(f1, Qb, Kb, n0, kc + 3, lane);
    }
    s_mma(acc, f0);
    s_mma(acc, f1);

    int mB = (lane < 16) ? 0 : 8;
#pragma unroll
    for (int t = 0; t < 4; ++t) {
      int nL = n0 + t * 16 + (lane & 15);
#pragma unroll
      for (int r = 0; r < 8; ++r)
        Ssc[(mB + r) * SEQ + nL] = acc[t][r] * 0.03125f;   // 1/sqrt(1024)
    }
  }
  __syncthreads();

  // ---- row softmax: 8 lanes per row, shuffle reductions ----
  {
    int row = threadIdx.x >> 3;                 // 0..15
    int sub = threadIdx.x & 7;
    float* rowp = Ssc + row * SEQ;
    float mx = -3.4e38f;
    for (int c = sub; c < SEQ; c += 8) mx = fmaxf(mx, rowp[c]);
    mx = fmaxf(mx, __shfl_xor(mx, 1, 8));
    mx = fmaxf(mx, __shfl_xor(mx, 2, 8));
    mx = fmaxf(mx, __shfl_xor(mx, 4, 8));
    float sum = 0.f;
    for (int c = sub; c < SEQ; c += 8) {
      float e = __expf(rowp[c] - mx);
      rowp[c] = e;
      sum += e;
    }
    sum += __shfl_xor(sum, 1, 8);
    sum += __shfl_xor(sum, 2, 8);
    sum += __shfl_xor(sum, 4, 8);
    float inv = 1.f / sum;
    bf16* prow = P + row * SEQ;
    for (int c = sub; c < SEQ; c += 8) prow[c] = (bf16)(rowp[c] * inv);
  }
  __syncthreads();

  // ---- O = P @ V  (Vt rows == Bt layout); wave owns 16 d-tiles ----
  // Ping-pong the cross-kc carried fragments (A from LDS, B tile0 from L2).
  float* Ob = O + (size_t)b * SEQ * DMODEL + (size_t)m0 * DMODEL;
  f32x8 acc[16] = {};
  const bf16* Vb0 = Vb + (size_t)(wave * 16) * 16 * SEQ;

  bf16x16 aX  = load_frag_A(P, SEQ, lane);
  bf16x16 b0X = load_frag_B(Vb0, SEQ, lane);
  bf16x16 aY  = load_frag_A(P + 32, SEQ, lane);
  bf16x16 b0Y = load_frag_B(Vb0 + 32, SEQ, lane);

  for (int kc = 0; kc < SEQ / 32 - 2; kc += 2) {
    pv_step(acc, aX, b0X, Vb, wave, kc, lane);
    aX  = load_frag_A(P + (kc + 2) * 32, SEQ, lane);
    b0X = load_frag_B(Vb0 + (kc + 2) * 32, SEQ, lane);
    pv_step(acc, aY, b0Y, Vb, wave, kc + 1, lane);
    aY  = load_frag_A(P + (kc + 3) * 32, SEQ, lane);
    b0Y = load_frag_B(Vb0 + (kc + 3) * 32, SEQ, lane);
  }
  pv_step(acc, aX, b0X, Vb, wave, SEQ / 32 - 2, lane);
  pv_step(acc, aY, b0Y, Vb, wave, SEQ / 32 - 1, lane);

  int nL = lane & 15;
  int mB = (lane < 16) ? 0 : 8;
#pragma unroll
  for (int t = 0; t < 16; ++t) {
    int n = (wave * 16 + t) * 16 + nL;
#pragma unroll
    for (int r = 0; r < 8; ++r)
      Ob[(size_t)(mB + r) * DMODEL + n] = acc[t][r];
  }
}

// ---------------------------------------------------------------------------
extern "C" void kernel_launch(void* const* d_in, const int* in_sizes, int n_in,
                              void* d_out, int out_size, void* d_ws, size_t ws_size,
                              hipStream_t stream) {
  const float* x  = (const float*)d_in[0];
  const float* Wq = (const float*)d_in[1];
  const float* bq = (const float*)d_in[2];
  const float* Wk = (const float*)d_in[3];
  const float* bk = (const float*)d_in[4];
  const float* Wv = (const float*)d_in[5];
  const float* bv = (const float*)d_in[6];
  float* out = (float*)d_out;

  char* ws = (char*)d_ws;
  bf16* Xb  = (bf16*)(ws);
  bf16* WtQ = (bf16*)(ws + (size_t)32  * (1 << 20));
  bf16* WtK = (bf16*)(ws + (size_t)34  * (1 << 20));
  bf16* WtV = (bf16*)(ws + (size_t)36  * (1 << 20));
  bf16* Qb  = (bf16*)(ws + (size_t)38  * (1 << 20));
  bf16* Kb  = (bf16*)(ws + (size_t)70  * (1 << 20));
  bf16* Vt  = (bf16*)(ws + (size_t)102 * (1 << 20));

  // 1) casts
  cast_x_kernel<<<(ROWS * DMODEL) / (256 * 4), 256, 0, stream>>>(x, Xb);
  cast_w_kernel<<<(DMODEL * DMODEL) / 256, 256, 0, stream>>>(Wq, WtQ);
  cast_w_kernel<<<(DMODEL * DMODEL) / 256, 256, 0, stream>>>(Wk, WtK);
  cast_w_kernel<<<(DMODEL * DMODEL) / 256, 256, 0, stream>>>(Wv, WtV);

  // 2) QKV projections (wave-tiles: ROWS/32 * DMODEL/64 = 8192; 8 waves/block)
  int gemmBlocks = (ROWS / 32) * (DMODEL / 64) / 8;   // 1024
  qkv_gemm_kernel<<<gemmBlocks, 256, 0, stream>>>(Xb, WtQ, bq, Qb, 0);
  qkv_gemm_kernel<<<gemmBlocks, 256, 0, stream>>>(Xb, WtK, bk, Kb, 0);
  qkv_gemm_kernel<<<gemmBlocks, 256, 0, stream>>>(Xb, WtV, bv, Vt, 1);

  // 3) fused attention: 8 batches * 128 query blocks; 192KB dynamic LDS
  size_t smemBytes = (size_t)MQ * SEQ * 4 + (size_t)MQ * SEQ * 2;  // 196608
  attn_kernel<<<BATCH * (SEQ / MQ), 128, smemBytes, stream>>>(Qb, Kb, Vt, out);
}
